// MinkUNet_24996709663121
// MI455X (gfx1250) — matrix operand order repaired and checked by reading, not compiled, not run on hardware
//
#include <hip/hip_runtime.h>

// ---------------------------------------------------------------------------
// MinkUNet (dense 2x4x64^3) on gfx1250.
// All matmuls (3^3/2^3/1^3 convs, deconv taps, classifier) run as implicit
// GEMM on v_wmma_f32_16x16x32_bf16. Activations/weights are bf16 (raw u16
// bits), accumulation + BatchNorm statistics are fp32.
// Per-wave tile: 32x32 output (4 WMMAs, A/B fragment reuse), block tile
// 256(M) x 32(N), K staged through LDS in chunks of 32 with 16B-aligned
// padded rows so fragments load as ds_load_b128.
// ---------------------------------------------------------------------------

typedef unsigned short u16;
typedef __attribute__((ext_vector_type(16))) __bf16 v16bf;
typedef __attribute__((ext_vector_type(8)))  float  v8f;
typedef __attribute__((ext_vector_type(8)))  unsigned int v8u;
typedef __attribute__((ext_vector_type(4)))  unsigned int v4u;

__device__ __forceinline__ u16 f2bf(float f) {
    unsigned u = __builtin_bit_cast(unsigned, f);
    unsigned r = u + 0x7FFFu + ((u >> 16) & 1u);   // round-to-nearest-even
    return (u16)(r >> 16);
}
__device__ __forceinline__ float bf2f(u16 h) {
    unsigned u = ((unsigned)h) << 16;
    return __builtin_bit_cast(float, u);
}

// A fragment (ISA 7.12.2, 16-bit A 16x32): lane half 0 -> K {0..7,16..23},
// half 1 -> K {8..15,24..31}; two 16B LDS loads.
__device__ __forceinline__ v16bf ld_fragA(const u16* p, int half) {
    v4u a0 = *(const v4u*)(p + half * 8);
    v4u a1 = *(const v4u*)(p + 16 + half * 8);
    v8u u;
    u[0] = a0[0]; u[1] = a0[1]; u[2] = a0[2]; u[3] = a0[3];
    u[4] = a1[0]; u[5] = a1[1]; u[6] = a1[2]; u[7] = a1[3];
    return __builtin_bit_cast(v16bf, u);
}
// B fragment: same K order as A: lane half 0 -> K {0..7,16..23},
// half 1 -> K {8..15,24..31}; two 16B LDS loads.
__device__ __forceinline__ v16bf ld_fragB(const u16* p, int half) {
    v4u b0 = *(const v4u*)(p + half * 8);
    v4u b1 = *(const v4u*)(p + 16 + half * 8);
    v8u u;
    u[0] = b0[0]; u[1] = b0[1]; u[2] = b0[2]; u[3] = b0[3];
    u[4] = b1[0]; u[5] = b1[1]; u[6] = b1[2]; u[7] = b1[3];
    return __builtin_bit_cast(v16bf, u);
}

// --------------------------- fp32 -> bf16 convert ---------------------------
__global__ void mk_cvt_bf(const float* __restrict__ s, u16* __restrict__ d, long n) {
    long i = (long)blockIdx.x * blockDim.x + threadIdx.x;
    if (i < n) d[i] = f2bf(s[i]);
}

// --------------------------- skip copy into concat --------------------------
__global__ void mk_copy_ch(const u16* __restrict__ src, u16* __restrict__ dst,
                           int C, int SP, int Ctot, int coff, long N) {
    long i = (long)blockIdx.x * blockDim.x + threadIdx.x;
    if (i >= N) return;
    int sp = (int)(i % SP);
    long t = i / SP;
    int c = (int)(t % C);
    int b = (int)(t / C);
    dst[((size_t)b * Ctot + c + coff) * SP + sp] = src[i];
}

// --------------------------- BatchNorm reduce -------------------------------
// One block per channel; training-mode biased variance like x.var().
__global__ __launch_bounds__(256) void mk_bn_reduce(
        const float* __restrict__ Y, float* __restrict__ stats,
        int C, int SP, int NB) {
    int c = blockIdx.x;
    long n = (long)NB * SP;
    float s = 0.f, s2 = 0.f;
    for (long i = threadIdx.x; i < n; i += 256) {
        int b = (int)(i / SP);
        int sp = (int)(i - (long)b * SP);
        float v = Y[((size_t)b * C + c) * SP + sp];
        s += v; s2 += v * v;
    }
    __shared__ float rs[256], rq[256];
    rs[threadIdx.x] = s; rq[threadIdx.x] = s2;
    __syncthreads();
    for (int st = 128; st > 0; st >>= 1) {
        if (threadIdx.x < (unsigned)st) {
            rs[threadIdx.x] += rs[threadIdx.x + st];
            rq[threadIdx.x] += rq[threadIdx.x + st];
        }
        __syncthreads();
    }
    if (threadIdx.x == 0) {
        float m = rs[0] / (float)n;
        float var = rq[0] / (float)n - m * m;
        stats[2 * c] = m;
        stats[2 * c + 1] = rsqrtf(var + 1e-5f);
    }
}

// --------------------------- BatchNorm apply --------------------------------
__global__ void mk_bn_apply(const float* __restrict__ Y, const float* __restrict__ stats,
                            const float* __restrict__ g, const float* __restrict__ bt,
                            const u16* __restrict__ res, u16* __restrict__ out,
                            int C, int SP, int Ctot, int coff, int relu, long N) {
    long i = (long)blockIdx.x * blockDim.x + threadIdx.x;
    if (i >= N) return;
    int sp = (int)(i % SP);
    long t = i / SP;
    int c = (int)(t % C);
    int b = (int)(t / C);
    float v = (Y[i] - stats[2 * c]) * stats[2 * c + 1] * g[c] + bt[c];
    if (res) v += bf2f(res[i]);
    if (relu) v = fmaxf(v, 0.f);
    out[((size_t)b * Ctot + c + coff) * SP + sp] = f2bf(v);
}

// --------------------------- WMMA implicit-GEMM conv ------------------------
// GEMM: M = NB*Do^3 output voxels, N = O channels, K = Cin*ks^3.
// Block: 256 thr = 8 waves. Tile: 256(M) x 32(N); wave tile 32x32 = 4 WMMAs.
// mode 0: normal conv store [NB,O,Do^3]
// mode 1: deconv tap scatter to [NB,O,(2Do)^3], tap mirrored (JAX conv_transpose)
// mode 2: row-major [M,O] store + bias (classifier)
__global__ __launch_bounds__(256) void mk_conv_wmma(
        const u16* __restrict__ X, const u16* __restrict__ Wt,
        float* __restrict__ Y, const float* __restrict__ bias,
        int NB, int Cin, int Di, int O, int Do,
        int ks, int stride, int pad,
        int Ktot, int wNs, int wKs, int wOff,
        int mode, int tap, int M) {
    (void)NB;
    __shared__ u16 As[256 * 40];   // 256 rows (M) x 32 K, padded to 40 (80B, 16B-aligned)
    __shared__ u16 Bs[32 * 40];    // 32 cols (N) x 32 K, padded to 40

    const int SPi = Di * Di * Di;
    const int SPo = Do * Do * Do;
    const int ks2 = ks * ks, ks3 = ks2 * ks;

    const int tid = threadIdx.x;
    const int mbase = blockIdx.x * 256;
    const int nbase = blockIdx.y * 32;

    // ---- per-thread A-staging coordinates: one thread per M row, 32 K each.
    // Consecutive lanes -> consecutive x voxels -> coalesced global loads.
    const int m = mbase + tid;
    const int mr = (m < M) ? m : (M - 1);          // clamp: no OOB reads
    const bool mvalid = (m < M);
    const int b = mr / SPo;
    const int sp = mr - b * SPo;
    const int oz = sp / (Do * Do);
    const int rrem = sp - oz * Do * Do;
    const int oy = rrem / Do;
    const int ox = rrem - oy * Do;
    const int bz = oz * stride - pad, by = oy * stride - pad, bx = ox * stride - pad;
    const u16* Xb = X + (size_t)b * Cin * SPi;
    const int spin = (oz * Di + oy) * Di + ox;      // ks==1 path (stride 1, pad 0)

    // ---- per-thread B-staging coordinates (8 threads per N col, 4 K each)
    const int nl = tid >> 3;
    const int kk0 = (tid & 7) * 4;
    const int nglob = nbase + nl;
    const u16* wrow = Wt + (size_t)nglob * wNs + wOff;

    // ---- fragment lane mapping (ISA 7.12.2)
    const int wave = tid >> 5;
    const int lane = tid & 31;
    const int half = lane >> 4;
    const int lo = lane & 15;
    const u16* arp0 = &As[(wave * 32 + lo) * 40];
    const u16* arp1 = &As[(wave * 32 + 16 + lo) * 40];
    const u16* brp0 = &Bs[lo * 40];
    const u16* brp1 = &Bs[(16 + lo) * 40];

    v8f acc00 = {}, acc01 = {}, acc10 = {}, acc11 = {};
    const int nK = (Ktot + 31) >> 5;

    for (int kc = 0; kc < nK; ++kc) {
        const int kb = kc * 32;
        // ---------------- stage A row (32 K values for this thread's voxel) --
        {
            u16* dst = &As[tid * 40];
            if (ks == 1) {
#pragma unroll 8
                for (int j = 0; j < 32; ++j) {
                    int k = kb + j;
                    dst[j] = (mvalid && k < Ktot) ? Xb[(size_t)k * SPi + spin] : (u16)0;
                }
            } else {
                int ci = kb / ks3;
                int t = kb - ci * ks3;
                int kd = t / ks2;
                int tr = t - kd * ks2;
                int kh = tr / ks;
                int kw = tr - kh * ks;
#pragma unroll 4
                for (int j = 0; j < 32; ++j) {
                    u16 v = 0;
                    if (mvalid && kb + j < Ktot) {
                        int iz = bz + kd, iy = by + kh, ix = bx + kw;
                        if ((unsigned)iz < (unsigned)Di && (unsigned)iy < (unsigned)Di &&
                            (unsigned)ix < (unsigned)Di)
                            v = Xb[(size_t)ci * SPi + (iz * Di + iy) * Di + ix];
                    }
                    dst[j] = v;
                    ++kw;
                    if (kw == ks) { kw = 0; ++kh;
                        if (kh == ks) { kh = 0; ++kd;
                            if (kd == ks) { kd = 0; ++ci; } } }
                }
            }
        }
        // ---------------- stage B tile ----------------
        {
#pragma unroll
            for (int j = 0; j < 4; ++j) {
                int k = kb + kk0 + j;
                u16 v = 0;
                if (k < Ktot && nglob < O) v = wrow[(size_t)k * wKs];
                Bs[nl * 40 + kk0 + j] = v;
            }
            int kpf = kb + 32 + kk0;
            if (kpf < Ktot && nglob < O)
                __builtin_prefetch(&wrow[(size_t)kpf * wKs], 0, 1);
        }
        __syncthreads();

        // ---------------- fragments + 4 WMMAs (A/B reuse) ----------------
        v16bf a0 = ld_fragA(arp0, half);
        v16bf a1 = ld_fragA(arp1, half);
        v16bf b0 = ld_fragB(brp0, half);
        v16bf b1 = ld_fragB(brp1, half);
        acc00 = __builtin_amdgcn_wmma_f32_16x16x32_bf16(false, a0, false, b0,
                                                        (short)0, acc00, false, false);
        acc01 = __builtin_amdgcn_wmma_f32_16x16x32_bf16(false, a0, false, b1,
                                                        (short)0, acc01, false, false);
        acc10 = __builtin_amdgcn_wmma_f32_16x16x32_bf16(false, a1, false, b0,
                                                        (short)0, acc10, false, false);
        acc11 = __builtin_amdgcn_wmma_f32_16x16x32_bf16(false, a1, false, b1,
                                                        (short)0, acc11, false, false);
        __syncthreads();
    }

    // ---------------- epilogue store (C layout: N=lo, M=half*8+r) ----------
#pragma unroll
    for (int ms = 0; ms < 2; ++ms) {
#pragma unroll
        for (int ns = 0; ns < 2; ++ns) {
            v8f a = (ms == 0) ? ((ns == 0) ? acc00 : acc01)
                              : ((ns == 0) ? acc10 : acc11);
            int o = nbase + ns * 16 + lo;
            if (o >= O) continue;
            int mrow0 = mbase + wave * 32 + ms * 16 + half * 8;
#pragma unroll
            for (int r = 0; r < 8; ++r) {
                int mm = mrow0 + r;
                if (mm >= M) continue;
                float val = a[r];
                if (mode == 2) {
                    Y[(size_t)mm * O + o] = val + (bias ? bias[o] : 0.f);
                } else if (mode == 1) {
                    int bb = mm / SPo;
                    int s2 = mm - bb * SPo;
                    int zi = s2 / (Do * Do);
                    int r3 = s2 - zi * Do * Do;
                    int yi = r3 / Do;
                    int xi = r3 - yi * Do;
                    int kd = (tap >> 2) & 1, kh = (tap >> 1) & 1, kw = tap & 1;
                    int z = 2 * zi + (1 - kd), yy = 2 * yi + (1 - kh), xx = 2 * xi + (1 - kw);
                    int Df = 2 * Do;
                    size_t SPf = (size_t)Df * Df * Df;
                    Y[((size_t)bb * O + o) * SPf + ((size_t)z * Df + yy) * Df + xx] = val;
                } else {
                    int bb = mm / SPo;
                    int s2 = mm - bb * SPo;
                    Y[((size_t)bb * O + o) * SPo + s2] = val;
                }
            }
        }
    }
}

// ===========================================================================
// Host orchestration
// ===========================================================================
extern "C" void kernel_launch(void* const* d_in, const int* in_sizes, int n_in,
                              void* d_out, int out_size, void* d_ws, size_t ws_size,
                              hipStream_t stream) {
    (void)in_sizes; (void)n_in; (void)out_size;
    static const int CSc[9] = {32, 32, 64, 128, 256, 256, 128, 96, 96};
    static const int DL[5] = {64, 32, 16, 8, 4};
    const int NB = 2;

    // ---- unpack inputs (setup_inputs() dict insertion order, recursively) ----
    int cur = 0;
    auto nxt = [&]() { return (const float*)d_in[cur++]; };
    const float* xin = nxt();
    const float *sw1 = nxt(), *sg1 = nxt(), *sb1 = nxt();
    const float *sw2 = nxt(), *sg2 = nxt(), *sb2 = nxt();
    struct RP { const float *w1, *g1, *b1, *w2, *g2, *b2, *wd, *gd, *bd; bool hd; };
    auto rdRP = [&](bool hd) {
        RP p{};
        p.w1 = nxt(); p.g1 = nxt(); p.b1 = nxt();
        p.w2 = nxt(); p.g2 = nxt(); p.b2 = nxt();
        p.hd = hd;
        if (hd) { p.wd = nxt(); p.gd = nxt(); p.bd = nxt(); }
        return p;
    };
    struct EN { const float *dw, *dg, *db; RP r1, r2; } enc[4];
    for (int i = 0; i < 4; i++) {
        enc[i].dw = nxt(); enc[i].dg = nxt(); enc[i].db = nxt();
        enc[i].r1 = rdRP(CSc[i] != CSc[i + 1]);
        enc[i].r2 = rdRP(false);
    }
    struct DE { const float *uw, *ug, *ub; RP r1, r2; } dec[4];
    for (int i = 0; i < 4; i++) {
        dec[i].uw = nxt(); dec[i].ug = nxt(); dec[i].ub = nxt();
        dec[i].r1 = rdRP(true);
        dec[i].r2 = rdRP(false);
    }
    const float* cw = nxt();
    const float* cb = nxt();

    // ---- workspace layout (bump allocator) ----
    char* ws = (char*)d_ws;
    size_t off = 0;
    auto alloc = [&](size_t bytes) -> void* {
        size_t o = off;
        off = (off + bytes + 255) & ~(size_t)255;
        return (void*)(ws + o);
    };
    float* stats = (float*)alloc(4096);                                // per-channel mean/istd
    u16* wscr = (u16*)alloc((size_t)6 << 20);                          // bf16 weight scratch
    float* f32 = (float*)alloc((size_t)NB * 96 * 262144 * sizeof(float)); // conv fp32 out
    const size_t actcap = (size_t)NB * 128 * 262144 * sizeof(u16);     // 134 MB each
    u16* bufA = (u16*)alloc(actcap);
    u16* bufB = (u16*)alloc(actcap);
    u16* bufC = (u16*)alloc(actcap);
    u16* S[5];
    for (int i = 0; i < 5; i++)
        S[i] = (u16*)alloc((size_t)NB * CSc[i] * DL[i] * DL[i] * DL[i] * sizeof(u16));
    if (off > ws_size) return;  // insufficient scratch; nothing safe to do

    // ---- launch helpers ----
    auto cvt = [&](const float* s, u16* d, long n) {
        mk_cvt_bf<<<(unsigned)((n + 255) / 256), 256, 0, stream>>>(s, d, n);
    };
    auto conv = [&](const u16* X, const float* Wf, int Cin, int Din, int O,
                    int ks, int st, int pad, int Dout, float* Yout) {
        long wn = (long)O * Cin * ks * ks * ks;
        cvt(Wf, wscr, wn);
        int M = NB * Dout * Dout * Dout;
        int Kt = Cin * ks * ks * ks;
        dim3 g((unsigned)((M + 255) / 256), (unsigned)((O + 31) / 32));
        mk_conv_wmma<<<g, 256, 0, stream>>>(X, wscr, Yout, nullptr, NB, Cin, Din, O, Dout,
                                            ks, st, pad, Kt, Kt, 1, 0, 0, 0, M);
    };
    auto deconvL = [&](const u16* X, const float* Wf, int Cin, int Dh, int O, float* Yout) {
        cvt(Wf, wscr, (long)O * Cin * 8);
        int M = NB * Dh * Dh * Dh;
        dim3 g((unsigned)((M + 255) / 256), (unsigned)((O + 31) / 32));
        for (int t = 0; t < 8; t++)
            mk_conv_wmma<<<g, 256, 0, stream>>>(X, wscr, Yout, nullptr, NB, Cin, Dh, O, Dh,
                                                1, 1, 0, Cin, Cin * 8, 8, t, 1, t, M);
    };
    auto bn = [&](const float* Yin, const float* g_, const float* b_, int C, int Dd,
                  const u16* res, int relu, u16* outp, int Ctot, int coff) {
        int SP = Dd * Dd * Dd;
        mk_bn_reduce<<<(unsigned)C, 256, 0, stream>>>(Yin, stats, C, SP, NB);
        long N = (long)NB * C * SP;
        mk_bn_apply<<<(unsigned)((N + 255) / 256), 256, 0, stream>>>(
            Yin, stats, g_, b_, res, outp, C, SP, Ctot, coff, relu, N);
    };
    // Residual block: shortcut first so only one fp32 scratch is ever live.
    auto resblock = [&](const RP& p, const u16* xi, int Cin, int Dd, int O,
                        u16* outb, u16* tmpb) {
        const u16* sc;
        if (p.hd) {
            conv(xi, p.wd, Cin, Dd, O, 1, 1, 0, Dd, f32);
            bn(f32, p.gd, p.bd, O, Dd, nullptr, 0, outb, O, 0);
            sc = outb;
        } else {
            sc = xi;
        }
        conv(xi, p.w1, Cin, Dd, O, 3, 1, 1, Dd, f32);
        bn(f32, p.g1, p.b1, O, Dd, nullptr, 1, tmpb, O, 0);
        conv(tmpb, p.w2, O, Dd, O, 3, 1, 1, Dd, f32);
        bn(f32, p.g2, p.b2, O, Dd, sc, 1, outb, O, 0);
    };

    // ---------------- stem ----------------
    cvt(xin, bufA, (long)NB * 4 * 262144);
    conv(bufA, sw1, 4, 64, 32, 3, 1, 1, 64, f32);
    bn(f32, sg1, sb1, 32, 64, nullptr, 1, bufB, 32, 0);
    conv(bufB, sw2, 32, 64, 32, 3, 1, 1, 64, f32);
    bn(f32, sg2, sb2, 32, 64, nullptr, 1, S[0], 32, 0);

    // ---------------- encoder ----------------
    for (int i = 0; i < 4; i++) {
        int Ci = CSc[i], Co = CSc[i + 1], Dd = DL[i + 1];
        conv(S[i], enc[i].dw, Ci, DL[i], Ci, 2, 2, 0, Dd, f32);
        bn(f32, enc[i].dg, enc[i].db, Ci, Dd, nullptr, 1, bufB, Ci, 0);
        resblock(enc[i].r1, bufB, Ci, Dd, Co, bufA, bufC);
        resblock(enc[i].r2, bufA, Co, Dd, Co, S[i + 1], bufC);
    }

    // ---------------- decoder ----------------
    const u16* y = S[4];
    int Cy = CSc[4];
    for (int i = 0; i < 4; i++) {
        int Co = CSc[5 + i];
        int Dh = DL[4 - i];        // y (input) dims
        int Df = DL[3 - i];        // upsampled dims
        int Csk = CSc[3 - i];      // skip channels
        int Ctot = Co + Csk;
        u16* cat = (i & 1) ? bufB : bufA;
        u16* o1  = (i & 1) ? bufA : bufB;
        deconvL(y, dec[i].uw, Cy, Dh, Co, f32);
        bn(f32, dec[i].ug, dec[i].ub, Co, Df, nullptr, 1, cat, Ctot, 0);
        long Nsk = (long)NB * Csk * Df * Df * Df;
        mk_copy_ch<<<(unsigned)((Nsk + 255) / 256), 256, 0, stream>>>(
            S[3 - i], cat, Csk, Df * Df * Df, Ctot, Co, Nsk);
        resblock(dec[i].r1, cat, Ctot, Df, Co, o1, bufC);
        resblock(dec[i].r2, o1, Co, Df, Co, cat, bufC);
        y = cat;
        Cy = Co;
    }

    // ---------------- classifier: [524288,96] @ [96,19]^T + cb -> d_out ----
    cvt(cw, wscr, (long)19 * 96);
    int M = NB * 262144;
    mk_conv_wmma<<<dim3((unsigned)((M + 255) / 256), 1), 256, 0, stream>>>(
        y, wscr, (float*)d_out, cb, NB, 96, 64, 19, 64,
        1, 1, 0, 96, 96, 1, 0, 2, 0, M);
}
